// NSNetLayer_27144193311177
// MI455X (gfx1250) — compile-verified, matrix operand order checked by reading, not run on hardware
//
#include <hip/hip_runtime.h>
#include <math.h>

#define EMB       64
#define NUM_NODE  200000
#define NUM_EDGE  1000000
#define HALF_EDGE (NUM_EDGE / 2)
#define NTILES    (HALF_EDGE / 16)   // 31250 tiles of 16 edges

typedef __bf16 bf16;
typedef bf16  v16bf __attribute__((ext_vector_type(16)));
typedef float v8f   __attribute__((ext_vector_type(8)));

union FragBF { unsigned int u[8]; v16bf v; };

// ---------------- helpers ----------------

__device__ inline unsigned short f32_to_bf16_bits(float f) {
  unsigned int u = __float_as_uint(f);
  u += 0x7FFFu + ((u >> 16) & 1u);          // round-to-nearest-even
  return (unsigned short)(u >> 16);
}
__device__ inline unsigned int pack_bf16x2(float lo, float hi) {
  return (unsigned int)f32_to_bf16_bits(lo) |
         ((unsigned int)f32_to_bf16_bits(hi) << 16);
}

// Order-preserving float <-> uint mapping for atomicMax on f32
#define ORD_NEG_INF 0x007FFFFFu
__device__ inline unsigned int ord_of_float(float f) {
  unsigned int u = __float_as_uint(f);
  return (u & 0x80000000u) ? ~u : (u | 0x80000000u);
}
__device__ inline float float_of_ord(unsigned int u) {
  unsigned int b = (u & 0x80000000u) ? (u & 0x7FFFFFFFu) : ~u;
  return __uint_as_float(b);
}

// A fragment (16x32 bf16, M=lane&15 (^rx for flip), K pattern per CDNA5 ISA 7.12.2)
__device__ inline FragBF load_a_frag(const unsigned short* x, int kt, int lane, int rx) {
  FragBF f;
  int m = (lane & 15) ^ rx;
  int half = lane >> 4;
  const unsigned int* x32 = (const unsigned int*)x;
#pragma unroll
  for (int v = 0; v < 8; ++v) {
    int k = kt * 32 + (v >> 2) * 16 + half * 8 + (v & 3) * 2;
    f.u[v] = x32[(m * 64 + k) >> 1];
  }
  return f;
}

// B fragment (32x16 bf16) from LDS weights stored transposed [n][k], row length Kw
__device__ inline FragBF load_b_frag(const unsigned short* wT, int Kw, int nt, int kt, int lane) {
  FragBF f;
  int n = nt * 16 + (lane & 15);
  int half = lane >> 4;
  const unsigned int* w32 = (const unsigned int*)wT;
#pragma unroll
  for (int v = 0; v < 8; ++v) {
    int k = kt * 32 + (v >> 2) * 16 + half * 8 + (v & 3) * 2;
    f.u[v] = w32[(n * Kw + k) >> 1];
  }
  return f;
}

__device__ inline v8f wmma_bf16(const FragBF& a, const FragBF& b, v8f c) {
  return __builtin_amdgcn_wmma_f32_16x16x32_bf16(
      false, a.v, false, b.v, (short)0, c, false, false);
}

// D (lane=N, vgpr r -> M=r+8*half) back to row-major bf16 LDS tile
__device__ inline void store_d_lds(unsigned short* y, v8f acc, int nt, int lane, bool relu) {
  int n = nt * 16 + (lane & 15);
  int half = lane >> 4;
#pragma unroll
  for (int r = 0; r < 8; ++r) {
    float v = acc[r];
    if (relu) v = fmaxf(v, 0.f);
    y[(r + 8 * half) * 64 + n] = f32_to_bf16_bits(v);
  }
}

// ---------------- elementwise / segment kernels ----------------

__global__ void init_kernel(float* aggr, unsigned int* mmap, float* ssum, int n) {
  int i = blockIdx.x * blockDim.x + threadIdx.x;
  if (i < n) { aggr[i] = 0.f; mmap[i] = ORD_NEG_INF; ssum[i] = 0.f; }
}

__global__ void scatter_add_kernel(const int* __restrict__ edge_index,
                                   const float* __restrict__ emb,
                                   float* __restrict__ aggr) {
  int t = blockIdx.x * blockDim.x + threadIdx.x;   // NUM_EDGE*16 threads
  int e = t >> 4, c = (t & 15) * 4;
  if (e >= NUM_EDGE) return;
  int d = edge_index[e * 2 + 1];
  float4 v = *(const float4*)(emb + (size_t)e * 64 + c);
  float* a = aggr + (size_t)d * 64 + c;
  atomicAdd(a + 0, v.x); atomicAdd(a + 1, v.y);
  atomicAdd(a + 2, v.z); atomicAdd(a + 3, v.w);
}

__global__ void scatter_max_kernel(const int* __restrict__ edge_index,
                                   const int* __restrict__ c2l_index,
                                   const float* __restrict__ emb,
                                   unsigned int* __restrict__ mmap) {
  int t = blockIdx.x * blockDim.x + threadIdx.x;   // HALF_EDGE*16 threads
  int i = t >> 4, c = (t & 15) * 4;
  if (i >= HALF_EDGE) return;
  int eid = c2l_index[i];
  int s = edge_index[eid * 2 + 0];
  int d = edge_index[eid * 2 + 1];
  float4 v = *(const float4*)(emb + (size_t)d * 64 + c);
  unsigned int* mp = mmap + (size_t)s * 64 + c;
  atomicMax(mp + 0, ord_of_float(v.x)); atomicMax(mp + 1, ord_of_float(v.y));
  atomicMax(mp + 2, ord_of_float(v.z)); atomicMax(mp + 3, ord_of_float(v.w));
}

__global__ void ssum_kernel(const int* __restrict__ edge_index,
                            const int* __restrict__ c2l_index,
                            const float* __restrict__ emb,
                            const unsigned int* __restrict__ mmap,
                            float* __restrict__ ssum) {
  int t = blockIdx.x * blockDim.x + threadIdx.x;
  int i = t >> 4, c = (t & 15) * 4;
  if (i >= HALF_EDGE) return;
  int eid = c2l_index[i];
  int s = edge_index[eid * 2 + 0];
  int d = edge_index[eid * 2 + 1];
  float4 v = *(const float4*)(emb + (size_t)d * 64 + c);
  const unsigned int* mp = mmap + (size_t)s * 64 + c;
  float* sp = ssum + (size_t)s * 64 + c;
  atomicAdd(sp + 0, expf(v.x - float_of_ord(mp[0])));
  atomicAdd(sp + 1, expf(v.y - float_of_ord(mp[1])));
  atomicAdd(sp + 2, expf(v.z - float_of_ord(mp[2])));
  atomicAdd(sp + 3, expf(v.w - float_of_ord(mp[3])));
}

__global__ void lse_kernel(unsigned int* mmap, const float* __restrict__ ssum, int n) {
  int i = blockIdx.x * blockDim.x + threadIdx.x;
  if (i >= n) return;
  unsigned int u = mmap[i];
  float m = (u == ORD_NEG_INF) ? 0.f : float_of_ord(u);  // where(isfinite(m), m, 0)
  float s = ssum[i];
  ((float*)mmap)[i] = logf(s > 0.f ? s : 1.f) + m;       // reuse mmap as lse (f32)
}

// ---------------- WMMA 3-layer MLP over gathered rows ----------------
// x = P[src] - Q[dst]; 3 x (64x64) layers, relu after layers 0,1.
// MODE 0: write f32 to outF at rows idx[i].  MODE 1: write bf16 to outH at local rows.
template <int MODE>
__global__ __launch_bounds__(256) void mlp3_kernel(
    const int* __restrict__ edge_index, const int* __restrict__ idx,
    const float* __restrict__ P, const float* __restrict__ Q,
    const float* __restrict__ W, const float* __restrict__ B,
    float* __restrict__ outF, unsigned short* __restrict__ outH) {
  __shared__ unsigned short wT[3][64 * 64];   // transposed [n][k], bf16
  __shared__ float bias[3 * 64];
  __shared__ unsigned short xbuf[8][16 * 64]; // per-wave 16x64 bf16 tile

  int tid = threadIdx.x;
  for (int i = tid; i < 3 * 64 * 64; i += blockDim.x) {
    int layer = i >> 12, r = i & 4095, k = r >> 6, n = r & 63;
    wT[layer][n * 64 + k] = f32_to_bf16_bits(W[i]);
  }
  for (int i = tid; i < 3 * 64; i += blockDim.x) bias[i] = B[i];
  __syncthreads();

  int lane = tid & 31, wave = tid >> 5, half = lane >> 4;
  unsigned short* xb = xbuf[wave];
  int nWaves = gridDim.x * (blockDim.x >> 5);
  for (int tile = blockIdx.x * (blockDim.x >> 5) + wave; tile < NTILES; tile += nWaves) {
    int base = tile * 16;
    {   // build input tile: lane handles row lane>>1, 32 channels
      int m = lane >> 1, ch0 = (lane & 1) * 32;
      int eid = idx[base + m];
      int s = edge_index[eid * 2 + 0];
      int d = edge_index[eid * 2 + 1];
      const float4* p4 = (const float4*)(P + (size_t)s * 64 + ch0);
      const float4* q4 = (const float4*)(Q + (size_t)d * 64 + ch0);
      unsigned int* x32 = (unsigned int*)xb;
#pragma unroll
      for (int j = 0; j < 8; ++j) {
        float4 a = p4[j], b4 = q4[j];
        int c = m * 64 + ch0 + j * 4;
        x32[(c >> 1) + 0] = pack_bf16x2(a.x - b4.x, a.y - b4.y);
        x32[(c >> 1) + 1] = pack_bf16x2(a.z - b4.z, a.w - b4.w);
      }
    }
    asm volatile("s_wait_dscnt 0" ::: "memory");

    int orow[8] = {0};
    if (MODE == 0) {
#pragma unroll
      for (int r = 0; r < 8; ++r) orow[r] = idx[base + r + 8 * half];
    }

#pragma unroll
    for (int layer = 0; layer < 3; ++layer) {
      FragBF A0 = load_a_frag(xb, 0, lane, 0);
      FragBF A1 = load_a_frag(xb, 1, lane, 0);
      asm volatile("" ::: "memory");
#pragma unroll
      for (int nt = 0; nt < 4; ++nt) {
        FragBF B0 = load_b_frag(wT[layer], 64, nt, 0, lane);
        FragBF B1 = load_b_frag(wT[layer], 64, nt, 1, lane);
        float bv = bias[layer * 64 + nt * 16 + (lane & 15)];
        v8f acc;
#pragma unroll
        for (int r = 0; r < 8; ++r) acc[r] = bv;
        acc = wmma_bf16(A0, B0, acc);
        acc = wmma_bf16(A1, B1, acc);
        if (layer < 2) {
          store_d_lds(xb, acc, nt, lane, true);
        } else if (MODE == 0) {
          int n = nt * 16 + (lane & 15);
#pragma unroll
          for (int r = 0; r < 8; ++r) outF[(size_t)orow[r] * 64 + n] = acc[r];
        } else {
          int n = nt * 16 + (lane & 15);
#pragma unroll
          for (int r = 0; r < 8; ++r)
            outH[((size_t)base + r + 8 * half) * 64 + n] = f32_to_bf16_bits(acc[r]);
        }
      }
      asm volatile("s_wait_dscnt 0" ::: "memory");
    }
  }
}

// ---------------- merge: concat(msg, flip) @ W0 + 2-layer MLP ----------------
__global__ __launch_bounds__(256) void merge_kernel(
    const int* __restrict__ l2c_index, const unsigned short* __restrict__ msg,
    const float* __restrict__ W0, const float* __restrict__ B0,
    const float* __restrict__ Wm, const float* __restrict__ Bm,
    float* __restrict__ outF) {
  __shared__ unsigned short wT0[64 * 128];    // [n][k], k in [0,128)
  __shared__ unsigned short wTm[2][64 * 64];
  __shared__ float bias0[64];
  __shared__ float biasm[2 * 64];
  __shared__ unsigned short xbuf[8][16 * 64];

  int tid = threadIdx.x;
  for (int i = tid; i < 128 * 64; i += blockDim.x) {
    int k = i >> 6, n = i & 63;
    wT0[n * 128 + k] = f32_to_bf16_bits(W0[i]);
  }
  for (int i = tid; i < 2 * 64 * 64; i += blockDim.x) {
    int layer = i >> 12, r = i & 4095, k = r >> 6, n = r & 63;
    wTm[layer][n * 64 + k] = f32_to_bf16_bits(Wm[i]);
  }
  for (int i = tid; i < 64; i += blockDim.x) bias0[i] = B0[i];
  for (int i = tid; i < 128; i += blockDim.x) biasm[i] = Bm[i];
  __syncthreads();

  int lane = tid & 31, wave = tid >> 5, half = lane >> 4;
  unsigned short* xb = xbuf[wave];
  int nWaves = gridDim.x * (blockDim.x >> 5);
  for (int tile = blockIdx.x * (blockDim.x >> 5) + wave; tile < NTILES; tile += nWaves) {
    int base = tile * 16;
    {   // async-copy bf16 msg tile straight into LDS (64 B per lane)
      int m = lane >> 1, ch0 = (lane & 1) * 32;
      unsigned long long ga =
          (unsigned long long)(uintptr_t)(msg + ((size_t)(base + m)) * 64 + ch0);
      unsigned int la = (unsigned int)(uintptr_t)(xb + m * 64 + ch0);
      asm volatile(
          "global_load_async_to_lds_b128 %0, %1, off\n\t"
          "global_load_async_to_lds_b128 %0, %1, off offset:16\n\t"
          "global_load_async_to_lds_b128 %0, %1, off offset:32\n\t"
          "global_load_async_to_lds_b128 %0, %1, off offset:48"
          :: "v"(la), "v"(ga) : "memory");
      asm volatile("s_wait_asynccnt 0" ::: "memory");
    }

    int orow[8];
#pragma unroll
    for (int r = 0; r < 8; ++r) orow[r] = l2c_index[base + r + 8 * half];

    {   // merge0: 128 -> 64, relu; K-tiles 2,3 read flipped rows
      FragBF A[4];
      A[0] = load_a_frag(xb, 0, lane, 0);
      A[1] = load_a_frag(xb, 1, lane, 0);
      A[2] = load_a_frag(xb, 0, lane, 1);
      A[3] = load_a_frag(xb, 1, lane, 1);
      asm volatile("" ::: "memory");
#pragma unroll
      for (int nt = 0; nt < 4; ++nt) {
        float bv = bias0[nt * 16 + (lane & 15)];
        v8f acc;
#pragma unroll
        for (int r = 0; r < 8; ++r) acc[r] = bv;
#pragma unroll
        for (int kt = 0; kt < 4; ++kt) {
          FragBF Bf = load_b_frag(wT0, 128, nt, kt, lane);
          acc = wmma_bf16(A[kt], Bf, acc);
        }
        store_d_lds(xb, acc, nt, lane, true);
      }
      asm volatile("s_wait_dscnt 0" ::: "memory");
    }

#pragma unroll
    for (int layer = 0; layer < 2; ++layer) {
      FragBF A0 = load_a_frag(xb, 0, lane, 0);
      FragBF A1 = load_a_frag(xb, 1, lane, 0);
      asm volatile("" ::: "memory");
#pragma unroll
      for (int nt = 0; nt < 4; ++nt) {
        FragBF B0f = load_b_frag(wTm[layer], 64, nt, 0, lane);
        FragBF B1f = load_b_frag(wTm[layer], 64, nt, 1, lane);
        float bv = biasm[layer * 64 + nt * 16 + (lane & 15)];
        v8f acc;
#pragma unroll
        for (int r = 0; r < 8; ++r) acc[r] = bv;
        acc = wmma_bf16(A0, B0f, acc);
        acc = wmma_bf16(A1, B1f, acc);
        if (layer == 0) {
          store_d_lds(xb, acc, nt, lane, true);
        } else {
          int n = nt * 16 + (lane & 15);
#pragma unroll
          for (int r = 0; r < 8; ++r) outF[(size_t)orow[r] * 64 + n] = acc[r];
        }
      }
      asm volatile("s_wait_dscnt 0" ::: "memory");
    }
  }
}

// ---------------- launcher ----------------

extern "C" void kernel_launch(void* const* d_in, const int* in_sizes, int n_in,
                              void* d_out, int out_size, void* d_ws, size_t ws_size,
                              hipStream_t stream) {
  const int*   edge_index = (const int*)d_in[0];
  const int*   l2c_index  = (const int*)d_in[1];
  const int*   c2l_index  = (const int*)d_in[2];
  const float* emb        = (const float*)d_in[3];
  const float* l2c_W      = (const float*)d_in[4];
  const float* l2c_b      = (const float*)d_in[5];
  const float* c2l_W      = (const float*)d_in[6];
  const float* c2l_b      = (const float*)d_in[7];
  const float* merge_W0   = (const float*)d_in[8];
  const float* merge_b0   = (const float*)d_in[9];
  const float* merge_W    = (const float*)d_in[10];
  const float* merge_b    = (const float*)d_in[11];
  float* outF = (float*)d_out;

  const size_t NODEF = (size_t)NUM_NODE * EMB;   // 12.8M
  char* ws = (char*)d_ws;
  float*          aggr = (float*)ws;
  unsigned int*   mmap = (unsigned int*)(ws + NODEF * 4);
  float*          ssum = (float*)(ws + NODEF * 8);
  unsigned short* msg  = (unsigned short*)(ws + NODEF * 12);  // 500K x 64 bf16

  init_kernel<<<(int)((NODEF + 255) / 256), 256, 0, stream>>>(aggr, mmap, ssum, (int)NODEF);
  scatter_add_kernel<<<(NUM_EDGE * 16) / 256, 256, 0, stream>>>(edge_index, emb, aggr);
  scatter_max_kernel<<<(HALF_EDGE * 16) / 256, 256, 0, stream>>>(edge_index, c2l_index, emb, mmap);
  ssum_kernel<<<(HALF_EDGE * 16) / 256, 256, 0, stream>>>(edge_index, c2l_index, emb, mmap, ssum);
  lse_kernel<<<(int)((NODEF + 255) / 256), 256, 0, stream>>>(mmap, ssum, (int)NODEF);

  // default output = input embedding (rows get overwritten below)
  hipMemcpyAsync(d_out, emb, (size_t)NUM_EDGE * EMB * sizeof(float),
                 hipMemcpyDeviceToDevice, stream);

  // Fixed grid + grid-stride: each wave handles ~4 tiles, amortizing the
  // per-block LDS weight staging while keeping ~8K waves in flight.
  int blocks = 1024;
  mlp3_kernel<0><<<blocks, 256, 0, stream>>>(edge_index, c2l_index,
      (const float*)mmap, emb, c2l_W, c2l_b, outF, nullptr);
  mlp3_kernel<1><<<blocks, 256, 0, stream>>>(edge_index, l2c_index,
      aggr, emb, l2c_W, l2c_b, nullptr, msg);
  merge_kernel<<<blocks, 256, 0, stream>>>(l2c_index, msg,
      merge_W0, merge_b0, merge_W, merge_b, outF);
}
